// GRUCell_45896020525411
// MI455X (gfx1250) — compile-verified, matrix operand order checked by reading, not run on hardware
//
#include <hip/hip_runtime.h>
#include <cstdint>
#include <cstddef>

// Problem dims (fixed by reference)
#define B_ROWS 8192
#define IN_DIM 2048
#define H_DIM  4096
#define K_DIM  (IN_DIM + H_DIM)   // 6144
#define N3     (3 * H_DIM)        // 12288

typedef __attribute__((ext_vector_type(16))) __bf16 v16bf;
typedef __attribute__((ext_vector_type(8)))  __bf16 v8bf;
typedef __attribute__((ext_vector_type(8)))  float  v8f;
typedef __attribute__((ext_vector_type(4)))  float  f4;

// GEMM tiling
#define BLK_M 128
#define BLK_N 128
#define BLK_K 32
#define LDB   40   // bf16 tile row stride (elements): 80B rows, 16B-aligned, bank-skewed

__device__ __forceinline__ uint32_t lds_off(const void* p) {
  // LDS aperture uses addr[31:0] as the LDS byte address
  return (uint32_t)(uintptr_t)p;
}

// One 16-byte async global->LDS copy per lane (gfx1250 ASYNCcnt path)
__device__ __forceinline__ void async_copy_b128(uint32_t dst_lds, const void* src) {
  asm volatile("global_load_async_to_lds_b128 %0, %1, off"
               :: "v"(dst_lds), "v"((uint64_t)(uintptr_t)src)
               : "memory");
}
__device__ __forceinline__ void wait_async_le8() {
  asm volatile("s_wait_asynccnt %0" :: "i"(8) : "memory");
}
__device__ __forceinline__ void wait_async_0() {
  asm volatile("s_wait_asynccnt %0" :: "i"(0) : "memory");
}

// Issue one K-tile (A panel + W panel) of async b128 copies: 8 instrs/wave.
__device__ __forceinline__ void issue_tile(int k0, uint32_t aoff, uint32_t boff,
                                           const float* __restrict__ inputs,
                                           const float* __restrict__ state,
                                           const float* __restrict__ W,
                                           int row0, int col0, int tid) {
#pragma unroll
  for (int i = 0; i < 4; ++i) {
    int idx = tid + i * 256;          // 1024 b128 chunks per 128x32 tile
    int r   = idx >> 3;               // 0..127
    int c   = (idx & 7) << 2;         // 0,4,...,28
    int gk  = k0 + c;                 // uniform source per k-block (2048%32==0)
    const float* ap = (gk < IN_DIM)
        ? inputs + (size_t)(row0 + r) * IN_DIM + gk
        : state  + (size_t)(row0 + r) * H_DIM + (gk - IN_DIM);
    async_copy_b128(aoff + idx * 16, ap);
    const float* bp = W + (size_t)(col0 + r) * K_DIM + gk;
    async_copy_b128(boff + idx * 16, bp);
  }
}

// Split one f32 into truncated-bf16 hi and residual-bf16 lo (as raw u16 bits)
__device__ __forceinline__ void split1(float x, uint32_t& h, uint32_t& l) {
  union { float f; uint32_t u; } a; a.f = x;
  h = a.u >> 16;
  union { uint32_t u; float f; } hr; hr.u = h << 16;
  union { float f; uint32_t u; } b; b.f = x - hr.f;
  l = b.u >> 16;
}

// Convert a staged f32 tile to packed bf16 hi/lo tiles (each element once per block).
__device__ __forceinline__ void convert_tile(const float* __restrict__ stage,
                                             __bf16* __restrict__ thi,
                                             __bf16* __restrict__ tlo, int tid) {
#pragma unroll
  for (int j = 0; j < 4; ++j) {
    int idx = tid + j * 256;
    int r   = idx >> 3;
    int c   = (idx & 7) << 2;
    f4 v = *(const f4*)(stage + idx * 4);   // idx*4 == r*32 + c
    uint32_t h0, h1, h2, h3, l0, l1, l2, l3;
    split1(v.x, h0, l0); split1(v.y, h1, l1);
    split1(v.z, h2, l2); split1(v.w, h3, l3);
    uint2 hp; hp.x = h0 | (h1 << 16); hp.y = h2 | (h3 << 16);
    uint2 lp; lp.x = l0 | (l1 << 16); lp.y = l2 | (l3 << 16);
    *(uint2*)((char*)thi + ((size_t)r * LDB + c) * 2) = hp;
    *(uint2*)((char*)tlo + ((size_t)r * LDB + c) * 2) = lp;
  }
}

// Fragment load: 16x32 16-bit A/B layout (ISA 7.12.2). base = &tile[row*LDB + half*8].
// K groups {half*8..+7} and {16+half*8..+7} are contiguous -> two ds_load_b128.
__device__ __forceinline__ v16bf load_frag(const __bf16* base) {
  v8bf a = *(const v8bf*)base;
  v8bf b = *(const v8bf*)(base + 16);
  return __builtin_shufflevector(a, b, 0,1,2,3,4,5,6,7,8,9,10,11,12,13,14,15);
}

__global__ __launch_bounds__(256)
void gru_gemm_kernel(const float* __restrict__ inputs,
                     const float* __restrict__ state,
                     const float* __restrict__ W,
                     float* __restrict__ parts) {
  __shared__ float  sA[2][BLK_M * BLK_K];   // async-filled f32 staging, 2x16KB
  __shared__ float  sB[2][BLK_N * BLK_K];   // async-filled f32 staging, 2x16KB
  __shared__ __bf16 tAhi[BLK_M * LDB];      // 10KB each
  __shared__ __bf16 tAlo[BLK_M * LDB];
  __shared__ __bf16 tBhi[BLK_N * LDB];
  __shared__ __bf16 tBlo[BLK_N * LDB];

  // ---- supertile swizzle for L2 reuse: 32 row-panels x 24 col-panels ----
  const int SUP_M = 32, SUP_N = 24;          // (32+24)*3MB = 168MB <= 192MB L2
  const int tilesN = N3 / BLK_N;             // 96
  const int supCols = tilesN / SUP_N;        // 4
  const int perSup  = SUP_M * SUP_N;         // 768
  int bid    = blockIdx.x;
  int sup    = bid / perSup;
  int within = bid % perSup;
  int supM   = sup / supCols;
  int supN   = sup % supCols;
  int tm = supM * SUP_M + (within % SUP_M);
  int tn = supN * SUP_N + (within / SUP_M);
  const int row0 = tm * BLK_M;
  const int col0 = tn * BLK_N;

  const int tid    = threadIdx.x;
  const int lane   = tid & 31;
  const int wid    = tid >> 5;
  const int wave_m = wid & 3;      // 4 waves along M (32 rows each)
  const int wave_n = wid >> 2;     // 2 waves along N (64 cols each)
  const int half   = lane >> 4;
  const int lane16 = lane & 15;

  uint32_t aoff[2] = { lds_off(&sA[0][0]), lds_off(&sA[1][0]) };
  uint32_t boff[2] = { lds_off(&sB[0][0]), lds_off(&sB[1][0]) };

  v8f acc[2][4];
#pragma unroll
  for (int mt = 0; mt < 2; ++mt)
#pragma unroll
    for (int nt = 0; nt < 4; ++nt)
      acc[mt][nt] = (v8f){0.f,0.f,0.f,0.f,0.f,0.f,0.f,0.f};

  // prologue: async-load K-tile 0 into buffer 0
  issue_tile(0, aoff[0], boff[0], inputs, state, W, row0, col0, tid);

  const int nK = K_DIM / BLK_K;    // 192
  for (int ki = 0; ki < nK; ++ki) {
    int cur = ki & 1, nxt = cur ^ 1;
    // software pipeline: issue next tile before waiting on current
    if (ki + 1 < nK) {
      issue_tile((ki + 1) * BLK_K, aoff[nxt], boff[nxt],
                 inputs, state, W, row0, col0, tid);
      wait_async_le8();            // current tile's 8 copies done; next 8 in flight
    } else {
      wait_async_0();
    }
    __syncthreads();               // all waves' portions landed; prev frag reads done

    // one-time f32 -> packed bf16 hi/lo conversion (each element once per block)
    convert_tile(&sA[cur][0], tAhi, tAlo, tid);
    convert_tile(&sB[cur][0], tBhi, tBlo, tid);
    __syncthreads();

    // ---- fragments + 3-term bf16 split-product WMMA ----
    v16bf bhi[4], blo[4];
#pragma unroll
    for (int nt = 0; nt < 4; ++nt) {
      int boffr = (wave_n * 64 + nt * 16 + lane16) * LDB + half * 8;
      bhi[nt] = load_frag(tBhi + boffr);
      blo[nt] = load_frag(tBlo + boffr);
    }
#pragma unroll
    for (int mt = 0; mt < 2; ++mt) {
      int aoffr = (wave_m * 32 + mt * 16 + lane16) * LDB + half * 8;
      v16bf ahi = load_frag(tAhi + aoffr);
      v16bf alo = load_frag(tAlo + aoffr);
#pragma unroll
      for (int nt = 0; nt < 4; ++nt) {
        acc[mt][nt] = __builtin_amdgcn_wmma_f32_16x16x32_bf16(
            false, ahi, false, bhi[nt], (short)0, acc[mt][nt], false, false);
        acc[mt][nt] = __builtin_amdgcn_wmma_f32_16x16x32_bf16(
            false, ahi, false, blo[nt], (short)0, acc[mt][nt], false, false);
        acc[mt][nt] = __builtin_amdgcn_wmma_f32_16x16x32_bf16(
            false, alo, false, bhi[nt], (short)0, acc[mt][nt], false, false);
      }
    }
    __syncthreads();
  }

  // ---- epilogue: C 16x16 f32 layout: VGPR r, lanes0-15 M=r, lanes16-31 M=8+r
#pragma unroll
  for (int mt = 0; mt < 2; ++mt) {
#pragma unroll
    for (int nt = 0; nt < 4; ++nt) {
      int n     = col0 + wave_n * 64 + nt * 16 + lane16;
      int mbase = row0 + wave_m * 32 + mt * 16 + half * 8;
#pragma unroll
      for (int r = 0; r < 8; ++r)
        parts[(size_t)(mbase + r) * N3 + n] = acc[mt][nt][r];
    }
  }
}

// ---- kernel 2: bias + LayerNorm over 3H row + GRU gating, one block per row ----
__global__ __launch_bounds__(256)
void gru_ln_gate_kernel(const float* __restrict__ parts,
                        const float* __restrict__ state,
                        const float* __restrict__ bias,
                        const float* __restrict__ ln_scale,
                        const float* __restrict__ ln_bias,
                        float* __restrict__ out) {
  __shared__ float rowv[N3];      // 48 KB
  __shared__ float red[2][8];
  __shared__ float stats[2];

  const int row = blockIdx.x;
  const int tid = threadIdx.x;
  const float* prow = parts + (size_t)row * N3;

  float s = 0.f, sq = 0.f;
#pragma unroll
  for (int i = 0; i < 12; ++i) {
    int e  = (i * 256 + tid) * 4;
    f4 v   = *(const f4*)(prow + e);
    f4 bb  = *(const f4*)(bias + e);
    v += bb;
    *(f4*)(rowv + e) = v;
    s  += v.x + v.y + v.z + v.w;
    sq += v.x*v.x + v.y*v.y + v.z*v.z + v.w*v.w;
  }
#pragma unroll
  for (int off = 16; off > 0; off >>= 1) {
    s  += __shfl_xor(s,  off, 32);
    sq += __shfl_xor(sq, off, 32);
  }
  if ((tid & 31) == 0) { red[0][tid >> 5] = s; red[1][tid >> 5] = sq; }
  __syncthreads();
  if (tid == 0) {
    float ts = 0.f, tq = 0.f;
#pragma unroll
    for (int w = 0; w < 8; ++w) { ts += red[0][w]; tq += red[1][w]; }
    float mu  = ts / (float)N3;
    float var = tq / (float)N3 - mu * mu;
    stats[0] = mu;
    stats[1] = rsqrtf(var + 1e-5f);
  }
  __syncthreads();
  const float mu = stats[0], rstd = stats[1];

  const float* srow = state + (size_t)row * H_DIM;
  float* orow = out + (size_t)row * H_DIM;
#pragma unroll
  for (int i = 0; i < 4; ++i) {
    int j = (i * 256 + tid) * 4;
#pragma unroll
    for (int u = 0; u < 4; ++u) {
      int c = j + u;
      float rv = (rowv[c]             - mu) * rstd * ln_scale[c]             + ln_bias[c];
      float cv = (rowv[c + H_DIM]     - mu) * rstd * ln_scale[c + H_DIM]     + ln_bias[c + H_DIM];
      float uv = (rowv[c + 2 * H_DIM] - mu) * rstd * ln_scale[c + 2 * H_DIM] + ln_bias[c + 2 * H_DIM];
      float reset = 1.f / (1.f + __expf(-rv));
      float cand  = tanhf(reset * cv);
      float upd   = 1.f / (1.f + __expf(-(uv - 1.0f)));   // UPDATE_BIAS = -1
      orow[c] = upd * cand + (1.f - upd) * srow[c];
    }
  }
}

extern "C" void kernel_launch(void* const* d_in, const int* in_sizes, int n_in,
                              void* d_out, int out_size, void* d_ws, size_t ws_size,
                              hipStream_t stream) {
  (void)in_sizes; (void)n_in; (void)out_size; (void)ws_size;
  const float* inputs  = (const float*)d_in[0];
  const float* state   = (const float*)d_in[1];
  const float* W       = (const float*)d_in[2];
  const float* b       = (const float*)d_in[3];
  const float* ln_s    = (const float*)d_in[4];
  const float* ln_b    = (const float*)d_in[5];
  float* out   = (float*)d_out;
  float* parts = (float*)d_ws;   // needs 8192*12288*4 = 384 MB scratch

  const int nTiles = (B_ROWS / BLK_M) * (N3 / BLK_N);  // 6144
  gru_gemm_kernel<<<dim3(nTiles), dim3(256), 0, stream>>>(inputs, state, W, parts);
  gru_ln_gate_kernel<<<dim3(B_ROWS), dim3(256), 0, stream>>>(parts, state, b, ln_s, ln_b, out);
}
